// PeriDynamicAttentionTriton_43800076485144
// MI455X (gfx1250) — compile-verified, matrix operand order checked by reading, not run on hardware
//
#include <hip/hip_runtime.h>
#include <hip/hip_bf16.h>

typedef __attribute__((ext_vector_type(16))) __bf16 v16bf;
typedef __attribute__((ext_vector_type(8)))  __bf16 v8bf;
typedef __attribute__((ext_vector_type(8)))  float  v8f;
typedef int v4i_ll __attribute__((vector_size(16)));   // matches builtin's V4i param

#define BT   2048   // B*T
#define CDIM 1024
#define NDV  1280   // 256 (disp) + 1024 (val)
#define NHh  16
#define BD   16
#define HS   64
#define DELTA 64

#if defined(__has_builtin)
#if __has_builtin(__builtin_amdgcn_global_load_async_to_lds_b128)
#define USE_ASYNC_LDS 1
#endif
#endif

// builtin signature (from hipcc diagnostic): (v4i addrspace(1)*, v4i addrspace(3)*, imm, imm)
#define GPTR128(p) ((__attribute__((address_space(1))) v4i_ll*)((void*)(p)))
#define LPTR128(p) ((__attribute__((address_space(3))) v4i_ll*)((void*)(p)))

__device__ __forceinline__ void wait_async_lds() {
#ifdef USE_ASYNC_LDS
#if __has_builtin(__builtin_amdgcn_s_wait_asynccnt)
    __builtin_amdgcn_s_wait_asynccnt(0);
#else
    asm volatile("s_wait_asynccnt 0" ::: "memory");
#endif
#endif
}

// ---------------- cast fp32 -> bf16 ----------------
__global__ void cast_f32_bf16(const float* __restrict__ src, __bf16* __restrict__ dst, int n) {
    int i = blockIdx.x * blockDim.x + threadIdx.x;
    if (i < n) dst[i] = (__bf16)src[i];
}

__global__ void copy_f32(const float* __restrict__ src, float* __restrict__ dst, int n) {
    int i = blockIdx.x * blockDim.x + threadIdx.x;
    if (i < n) dst[i] = src[i];
}

// ---------------- bf16 WMMA GEMM:  C[m,n] = sum_k A[m,k]*W[n,k] + bias[n] ----------------
// A: M x K row-major bf16 ; W: N x K row-major bf16 ; C: M x N row-major f32
// M % 128 == 0, N % 128 == 0, K % 32 == 0
#define GBM 128
#define GBN 128
#define GBK 32
#define LDSA (GBK + 8)   // row pad (bf16 elems) to dodge bank conflicts

// fragment per ISA 16-bit 16x32 layout:
// elems 0..7 -> K = lh*8 .. lh*8+7 ; elems 8..15 -> K = 16+lh*8 ..
__device__ __forceinline__ v16bf load_frag(const __bf16* r, int lh) {
    v8bf lo = *(const v8bf*)&r[lh * 8];
    v8bf hi = *(const v8bf*)&r[16 + lh * 8];
    return __builtin_shufflevector(lo, hi, 0,1,2,3,4,5,6,7,8,9,10,11,12,13,14,15);
}

__launch_bounds__(256)
__global__ void gemm_bf16_wmma(const __bf16* __restrict__ A, const __bf16* __restrict__ Bw,
                               const float* __restrict__ bias, float* __restrict__ C,
                               int M, int N, int K) {
    __shared__ __align__(16) __bf16 As[2][GBM * LDSA];
    __shared__ __align__(16) __bf16 Bs[2][GBN * LDSA];
    const int tid  = threadIdx.x;
    const int bm   = blockIdx.y * GBM;
    const int bn   = blockIdx.x * GBN;
    const int w    = tid >> 5, lane = tid & 31;
    const int wm   = (w & 3) * 32;           // 4 waves along M  -> wave tile 32 x 64
    const int wn   = (w >> 2) * 64;          // 2 waves along N
    const int lr   = lane & 15;
    const int lh   = lane >> 4;

    // stage one 128x32 A tile + 128x32 B tile into LDS buffer `buf`
    auto stage = [&](int buf, int k0) {
        #pragma unroll
        for (int i = 0; i < 2; ++i) {
            int li  = tid + i * 256;
            int row = li >> 2, cg = (li & 3) * 8;
#ifdef USE_ASYNC_LDS
            __builtin_amdgcn_global_load_async_to_lds_b128(
                GPTR128(&A[(size_t)(bm + row) * K + k0 + cg]),
                LPTR128(&As[buf][row * LDSA + cg]), 0, 0);
            __builtin_amdgcn_global_load_async_to_lds_b128(
                GPTR128(&Bw[(size_t)(bn + row) * K + k0 + cg]),
                LPTR128(&Bs[buf][row * LDSA + cg]), 0, 0);
#else
            *(v8bf*)&As[buf][row * LDSA + cg] =
                *(const v8bf*)&A[(size_t)(bm + row) * K + k0 + cg];
            *(v8bf*)&Bs[buf][row * LDSA + cg] =
                *(const v8bf*)&Bw[(size_t)(bn + row) * K + k0 + cg];
#endif
        }
    };

    v8f acc[2][4] = {};
    int cur = 0;
    stage(0, 0);

    for (int k0 = 0; k0 < K; k0 += GBK) {
        wait_async_lds();        // my async stores into `cur` are complete
        __syncthreads();         // everyone's are; prior reads of `cur^1` retired
        if (k0 + GBK < K) stage(cur ^ 1, k0 + GBK);

        v16bf af[2], bfr[4];
        #pragma unroll
        for (int i = 0; i < 2; ++i)
            af[i] = load_frag(&As[cur][(wm + i * 16 + lr) * LDSA], lh);
        #pragma unroll
        for (int j = 0; j < 4; ++j)
            bfr[j] = load_frag(&Bs[cur][(wn + j * 16 + lr) * LDSA], lh);
        #pragma unroll
        for (int i = 0; i < 2; ++i)
            #pragma unroll
            for (int j = 0; j < 4; ++j)
                acc[i][j] = __builtin_amdgcn_wmma_f32_16x16x32_bf16(
                    false, af[i], false, bfr[j], (short)0, acc[i][j], false, false);
        cur ^= 1;
    }

    // epilogue: VGPR r holds M = r (lanes 0-15) / M = 8+r (lanes 16-31); N = lr
    #pragma unroll
    for (int i = 0; i < 2; ++i)
        #pragma unroll
        for (int j = 0; j < 4; ++j) {
            int n  = bn + wn + j * 16 + lr;
            float bv = bias[n];
            #pragma unroll
            for (int r = 0; r < 8; ++r) {
                int m = bm + wm + i * 16 + lh * 8 + r;
                C[(size_t)m * N + n] = acc[i][j][r] + bv;
            }
        }
}

// ---------------- P = disp @ W_fused.T  (per (b,t,h): 16 -> 32) ----------------
__global__ void compute_P(const float* __restrict__ disp /* BTx1280, disp = cols 0..255 */,
                          const float* __restrict__ W_fused /* 32x16 */,
                          float* __restrict__ P /* (BT*16*32) */) {
    int idx = blockIdx.x * blockDim.x + threadIdx.x;   // BT*16*32 = 1,048,576
    if (idx >= BT * NHh * 32) return;
    int k  = idx & 31;
    int rh = idx >> 5;                // bt*16 + h
    int bt = rh >> 4, h = rh & 15;
    const float* drow = disp + (size_t)bt * NDV + h * BD;
    const float* wrow = W_fused + k * BD;
    float s = 0.f;
    #pragma unroll
    for (int d = 0; d < BD; ++d) s += drow[d] * wrow[d];
    P[idx] = s;
}

// ---------------- pos_feat = rel_pos_emb @ W_pos.T  (64x16) ----------------
__global__ void compute_posfeat(const float* __restrict__ rel, const float* __restrict__ W_pos,
                                float* __restrict__ pf) {
    int idx = blockIdx.x * blockDim.x + threadIdx.x;   // 1024
    if (idx >= DELTA * BD) return;
    int d = idx & 15, j = idx >> 4;
    float s = 0.f;
    #pragma unroll
    for (int e = 0; e < BD; ++e) s += rel[j * BD + e] * W_pos[d * BD + e];
    pf[idx] = s;
}

__device__ __forceinline__ float gelu_exact(float z) {
    return 0.5f * z * (1.f + erff(z * 0.70710678118654752440f));
}

// ---------------- windowed attention: logits, softmax, weighted val sum ----------------
// one block = 4 consecutive t for one (b,h); 64 threads per t
__launch_bounds__(256)
__global__ void attn_window(const float* __restrict__ P,        // (BT*16*32)
                            const float* __restrict__ valbuf,   // cc + 256, ld NDV
                            const float* __restrict__ posf,     // 64x16
                            const float* __restrict__ b_fused,  // 32
                            const float* __restrict__ W_bond,   // 16
                            const float* __restrict__ b_bond,   // 1
                            const float* __restrict__ W_dmg,    // 16
                            const float* __restrict__ b_dmg,    // 1
                            __bf16* __restrict__ attout)        // BT x 1024 bf16
{
    __shared__ float lsw[4][DELTA];
    __shared__ float ptb[4][32];
    const int tid = threadIdx.x;
    const int tt  = tid >> 6;
    const int jj  = tid & 63;
    const int bid = blockIdx.x;              // B*NH*(T/4) = 8192
    const int b   = bid >> 12;
    const int rem = bid & 4095;
    const int h   = rem >> 8;
    const int t   = ((rem & 255) << 2) + tt;
    const int btt = b * 1024 + t;

    if (jj < 32) ptb[tt][jj] = P[((size_t)btt * NHh + h) * 32 + jj];
    __syncthreads();

    const int s = t - (DELTA - 1) + jj;
    float logit = -__builtin_inff();
    if (s >= 0) {
        const float* Ps = P + ((size_t)(b * 1024 + s) * NHh + h) * 32;
        float bond = 0.f, dmg = 0.f;
        #pragma unroll
        for (int d = 0; d < BD; ++d) {
            float z1 = Ps[d]      - ptb[tt][d]      + b_fused[d]      + posf[jj * BD + d];
            float z2 = Ps[16 + d] - ptb[tt][16 + d] + b_fused[16 + d];
            bond += gelu_exact(z1) * W_bond[d];
            dmg  += gelu_exact(z2) * W_dmg[d];
        }
        float damage = 1.f / (1.f + __expf(-(dmg + b_dmg[0])));
        logit = bond + b_bond[0] - 10.f * damage;
    }
    lsw[tt][jj] = logit;
    __syncthreads();

    float mx = -__builtin_inff();
    #pragma unroll 8
    for (int j = 0; j < DELTA; ++j) mx = fmaxf(mx, lsw[tt][j]);
    float se = 0.f;
    #pragma unroll 8
    for (int j = 0; j < DELTA; ++j) se += __expf(lsw[tt][j] - mx);
    float wgt = __expf(logit - mx) / se;
    __syncthreads();
    lsw[tt][jj] = wgt;
    __syncthreads();

    // phase 3: jj reinterpreted as channel within head
    float acc = 0.f;
    const float* vbase = valbuf + (size_t)(b * 1024) * NDV + h * HS + jj;
    const int j0 = (t >= DELTA - 1) ? 0 : (DELTA - 1 - t);
    for (int j = j0; j < DELTA; ++j) {
        int ss = t - (DELTA - 1) + j;
        acc += lsw[tt][j] * vbase[(size_t)ss * NDV];
    }
    attout[(size_t)btt * CDIM + h * HS + jj] = (__bf16)acc;
}

// ---------------- launch ----------------
extern "C" void kernel_launch(void* const* d_in, const int* in_sizes, int n_in,
                              void* d_out, int out_size, void* d_ws, size_t ws_size,
                              hipStream_t stream) {
    const float* x       = (const float*)d_in[0];
    const float* W_disp  = (const float*)d_in[1];
    const float* b_disp  = (const float*)d_in[2];
    const float* W_val   = (const float*)d_in[3];
    const float* b_val   = (const float*)d_in[4];
    const float* rel     = (const float*)d_in[5];
    const float* W_fused = (const float*)d_in[6];
    const float* b_fused = (const float*)d_in[7];
    const float* W_pos   = (const float*)d_in[8];
    const float* W_bond  = (const float*)d_in[9];
    const float* b_bond  = (const float*)d_in[10];
    const float* W_dmg   = (const float*)d_in[11];
    const float* b_dmg   = (const float*)d_in[12];
    const float* W_cproj = (const float*)d_in[13];
    const float* b_cproj = (const float*)d_in[14];
    float* out = (float*)d_out;

    // workspace carve-up (~28 MB)
    char* ws = (char*)d_ws;
    size_t off = 0;
    auto carve = [&](size_t bytes) -> char* {
        char* p = ws + off;
        off = (off + bytes + 255) & ~(size_t)255;
        return p;
    };
    __bf16* xbf     = (__bf16*)carve((size_t)BT * CDIM * 2);
    __bf16* wdv     = (__bf16*)carve((size_t)NDV * CDIM * 2);   // [W_disp; W_val]
    __bf16* wcp     = (__bf16*)carve((size_t)CDIM * CDIM * 2);
    float*  cc      = (float*)carve((size_t)BT * NDV * 4);      // disp | val
    float*  Pbuf    = (float*)carve((size_t)BT * NHh * 32 * 4);
    float*  posf    = (float*)carve((size_t)DELTA * BD * 4);
    float*  biascat = (float*)carve((size_t)NDV * 4);
    __bf16* attout  = (__bf16*)carve((size_t)BT * CDIM * 2);

    // casts
    cast_f32_bf16<<<(BT * CDIM + 255) / 256, 256, 0, stream>>>(x, xbf, BT * CDIM);
    cast_f32_bf16<<<(256 * CDIM + 255) / 256, 256, 0, stream>>>(W_disp, wdv, 256 * CDIM);
    cast_f32_bf16<<<(CDIM * CDIM + 255) / 256, 256, 0, stream>>>(W_val, wdv + (size_t)256 * CDIM, CDIM * CDIM);
    cast_f32_bf16<<<(CDIM * CDIM + 255) / 256, 256, 0, stream>>>(W_cproj, wcp, CDIM * CDIM);
    copy_f32<<<1, 256, 0, stream>>>(b_disp, biascat, 256);
    copy_f32<<<4, 256, 0, stream>>>(b_val, biascat + 256, CDIM);

    // GEMM 1: [disp | val] = x @ [W_disp; W_val].T + bias   (2048 x 1280 x 1024)
    gemm_bf16_wmma<<<dim3(NDV / GBN, BT / GBM), 256, 0, stream>>>(
        xbf, wdv, biascat, cc, BT, NDV, CDIM);

    // tiny projections
    compute_P<<<(BT * NHh * 32 + 255) / 256, 256, 0, stream>>>(cc, W_fused, Pbuf);
    compute_posfeat<<<(DELTA * BD + 255) / 256, 256, 0, stream>>>(rel, W_pos, posf);

    // windowed attention
    attn_window<<<2 * NHh * (1024 / 4), 256, 0, stream>>>(
        Pbuf, cc + 256, posf, b_fused, W_bond, b_bond, W_dmg, b_dmg, attout);

    // GEMM 2: out = attout @ W_cproj.T + b_cproj   (2048 x 1024 x 1024)
    gemm_bf16_wmma<<<dim3(CDIM / GBN, BT / GBM), 256, 0, stream>>>(
        attout, wcp, b_cproj, out, BT, CDIM, CDIM);
}
